// WindowAttention_44633300140585
// MI455X (gfx1250) — compile-verified, matrix operand order checked by reading, not run on hardware
//
#include <hip/hip_runtime.h>
#include <hip/hip_bf16.h>
#include <stdint.h>

#define WS    7
#define SHIFT 3
#define NH    6
#define DIM   192
#define HD    32
#define LTOK  49
#define LP    64
#define HG    56
#define NQKV  (3 * DIM)   // 576

typedef __attribute__((ext_vector_type(16))) __bf16 v16bf;
typedef __attribute__((ext_vector_type(8)))  float  v8f;

union FragU { uint4 u4[2]; v16bf v; };

__device__ __forceinline__ unsigned short f2bf(float f) {
  unsigned int u = __float_as_uint(f);
  u += 0x7FFFu + ((u >> 16) & 1u);   // round-to-nearest-even
  return (unsigned short)(u >> 16);
}

// Load one 16x32 bf16 WMMA fragment (A: rows=M, or B: rows=cols-of-B, i.e. W row-major).
// Layout per ISA 7.12.2: lane r=lane&15 -> row, g=lane>>4 selects K half-block;
// VGPR j<4 covers K (8g..8g+7), j>=4 covers K (16+8g..16+8g+7) -> two b128 loads.
__device__ __forceinline__ v16bf load_frag(const unsigned short* base, int tile_row,
                                           int stride, int kblk) {
  const int lane = threadIdx.x & 31;
  const int g = lane >> 4;
  const int r = lane & 15;
  const unsigned short* p = base + (size_t)(tile_row + r) * stride + kblk + 8 * g;
  FragU f;
  f.u4[0] = *(const uint4*)(p);
  f.u4[1] = *(const uint4*)(p + 16);
  return f.v;
}

__device__ __forceinline__ int shift_region(int g) {
  return (g < HG - WS) ? 0 : ((g < HG - SHIFT) ? 1 : 2);
}

__global__ void prep_weights(const float* __restrict__ qkv_w, const float* __restrict__ proj_w,
                             unsigned short* __restrict__ wqkv, unsigned short* __restrict__ wproj) {
  int i = blockIdx.x * 256 + threadIdx.x;
  if (i < NQKV * DIM) wqkv[i] = f2bf(qkv_w[i]);
  if (i < DIM * DIM)  wproj[i] = f2bf(proj_w[i]);
}

__global__ void prep_bias(const float* __restrict__ table, float* __restrict__ bias_g) {
  int t = blockIdx.x * 256 + threadIdx.x;
  if (t >= NH * LTOK * LTOK) return;
  int h = t / (LTOK * LTOK);
  int rem = t - h * LTOK * LTOK;
  int l = rem / LTOK, m = rem - l * LTOK;
  int a = l / WS - m / WS + (WS - 1);
  int b = l % WS - m % WS + (WS - 1);
  bias_g[t] = table[(a * (2 * WS - 1) + b) * NH + h];
}

__global__ __launch_bounds__(256)
void swin_window_attn(const float* __restrict__ x,
                      const unsigned short* __restrict__ wqkv,
                      const float* __restrict__ qkv_b,
                      const unsigned short* __restrict__ wproj,
                      const float* __restrict__ proj_b,
                      const float* __restrict__ bias_g,
                      float* __restrict__ out)
{
  extern __shared__ char smem[];
  // Region map (bytes):
  //   [0      , 24576) Q  bf16 [NH][LP][HD]
  //   [24576  , 49152) K  bf16 [NH][LP][HD]
  //   [49152  , 73728) Vt bf16 [NH][HD][LP]   (transposed so P*V B-frags are K-contiguous)
  //   [73728  , 98304) X  bf16 [LP][DIM]      -- dead after QKV; reused as S(f32)+P(bf16)
  //   [73728  , 90112)   S  f32 [LP][LP]
  //   [90112  , 98304)   P  bf16 [LP][LP]
  //   [98304  ,122880) O  bf16 [LP][DIM]
  unsigned short* lq = (unsigned short*)smem;
  unsigned short* lk = lq + NH * LP * HD;
  unsigned short* lv = lk + NH * LP * HD;
  unsigned short* lx = lv + NH * HD * LP;
  float*          ls = (float*)(smem + 73728);
  unsigned short* lp = (unsigned short*)(smem + 90112);
  unsigned short* lo = (unsigned short*)(smem + 98304);

  const int tid  = threadIdx.x;
  const int wave = tid >> 5;
  const int lane = tid & 31;
  const int g    = lane >> 4;

  const int blk = blockIdx.x;
  const int nb  = blk >> 6;             // batch index
  const int wi  = blk & 63;             // window index in 8x8 grid
  const int wh7 = (wi >> 3) * WS;       // rolled-frame row base
  const int ww7 = (wi & 7) * WS;        // rolled-frame col base

  // ---- Phase 0: rolled window -> LDS bf16 (rows >= 49 zero), zero P padding ----
  for (int e = tid; e < LP * DIM; e += 256) {
    int l = e / DIM, d = e - l * DIM;
    unsigned short v = 0;
    if (l < LTOK) {
      int r = l / WS, c = l - r * WS;
      int gh = wh7 + r + SHIFT; if (gh >= HG) gh -= HG;
      int gw = ww7 + c + SHIFT; if (gw >= HG) gw -= HG;
      v = f2bf(x[(((size_t)nb * HG + gh) * HG + gw) * DIM + d]);
    }
    lx[e] = v;
  }
  for (int e = tid; e < LP * LP; e += 256) lp[e] = 0;  // P padding stays 0 forever
  __syncthreads();

  const float scale = 0.17677669529663689f;  // 1/sqrt(32), folded into Q

  // ---- Phase 1: QKV = Xw * Wqkv^T + b  (4 x 36 tiles, K=6 steps) ----
  for (int t = wave; t < 4 * (NQKV / 16); t += 8) {
    int mT = t / (NQKV / 16);
    int nT = t % (NQKV / 16);
    v8f acc = {};
    for (int ks = 0; ks < DIM / 32; ++ks) {
      v16bf a  = load_frag(lx, mT * 16, DIM, ks * 32);
      v16bf bm = load_frag(wqkv, nT * 16, DIM, ks * 32);
      acc = __builtin_amdgcn_wmma_f32_16x16x32_bf16(false, a, false, bm, (short)0, acc, false, false);
    }
    int n = nT * 16 + (lane & 15);
    float bias = qkv_b[n];
    for (int j = 0; j < 8; ++j) {
      int l = mT * 16 + 8 * g + j;
      float v = acc[j] + bias;
      if (n < DIM) {
        int h = n >> 5, d = n & 31;
        lq[(h * LP + l) * HD + d] = f2bf(v * scale);
      } else if (n < 2 * DIM) {
        int nn = n - DIM; int h = nn >> 5, d = nn & 31;
        lk[(h * LP + l) * HD + d] = f2bf(v);
      } else {
        int nn = n - 2 * DIM; int h = nn >> 5, d = nn & 31;
        lv[(h * HD + d) * LP + l] = f2bf(v);    // transposed store
      }
    }
  }
  __syncthreads();

  // ---- Phase 2: per-head attention ----
  for (int h = 0; h < NH; ++h) {
    const unsigned short* qh = lq + h * LP * HD;
    const unsigned short* kh = lk + h * LP * HD;
    const unsigned short* vh = lv + h * HD * LP;

    // (a) S = Q K^T : 16 tiles, single K-step of 32
    for (int t = wave; t < 16; t += 8) {
      int mT = t >> 2, nT = t & 3;
      v16bf a  = load_frag(qh, mT * 16, HD, 0);
      v16bf bm = load_frag(kh, nT * 16, HD, 0);
      v8f acc = {};
      acc = __builtin_amdgcn_wmma_f32_16x16x32_bf16(false, a, false, bm, (short)0, acc, false, false);
      int n = nT * 16 + (lane & 15);
      for (int j = 0; j < 8; ++j) ls[(mT * 16 + 8 * g + j) * LP + n] = acc[j];
    }
    __syncthreads();

    // (b) softmax over keys 0..48 with rel-pos bias + shift mask
    if (tid < LTOK) {
      int r = tid;
      int qreg = shift_region(wh7 + r / WS) * 3 + shift_region(ww7 + r % WS);
      const float* brow = bias_g + ((size_t)h * LTOK + r) * LTOK;
      float mx = -1e30f;
      for (int m = 0; m < LTOK; ++m) {
        int mreg = shift_region(wh7 + m / WS) * 3 + shift_region(ww7 + m % WS);
        float v = ls[r * LP + m] + brow[m];
        if (mreg != qreg) v -= 100.0f;
        ls[r * LP + m] = v;
        mx = fmaxf(mx, v);
      }
      float sum = 0.0f;
      for (int m = 0; m < LTOK; ++m) {
        float e = __expf(ls[r * LP + m] - mx);
        ls[r * LP + m] = e;
        sum += e;
      }
      float inv = 1.0f / sum;
      for (int m = 0; m < LTOK; ++m)
        lp[r * LP + m] = f2bf(ls[r * LP + m] * inv);
    }
    __syncthreads();

    // (c) O_h = P V : 8 tiles, one per wave, K=2 steps of 32
    {
      int mT = wave >> 1, nT = wave & 1;
      v8f acc = {};
      for (int ks = 0; ks < 2; ++ks) {
        v16bf a  = load_frag(lp, mT * 16, LP, ks * 32);
        v16bf bm = load_frag(vh, nT * 16, LP, ks * 32);
        acc = __builtin_amdgcn_wmma_f32_16x16x32_bf16(false, a, false, bm, (short)0, acc, false, false);
      }
      int n = h * HD + nT * 16 + (lane & 15);
      for (int j = 0; j < 8; ++j)
        lo[(mT * 16 + 8 * g + j) * DIM + n] = f2bf(acc[j]);
    }
    __syncthreads();
  }

  // ---- Phase 3: out = O * Wproj^T + b, scatter with un-roll ----
  for (int t = wave; t < 4 * (DIM / 16); t += 8) {
    int mT = t / (DIM / 16);
    int nT = t % (DIM / 16);
    v8f acc = {};
    for (int ks = 0; ks < DIM / 32; ++ks) {
      v16bf a  = load_frag(lo, mT * 16, DIM, ks * 32);
      v16bf bm = load_frag(wproj, nT * 16, DIM, ks * 32);
      acc = __builtin_amdgcn_wmma_f32_16x16x32_bf16(false, a, false, bm, (short)0, acc, false, false);
    }
    int n = nT * 16 + (lane & 15);
    float bias = proj_b[n];
    for (int j = 0; j < 8; ++j) {
      int l = mT * 16 + 8 * g + j;
      if (l < LTOK) {
        int r = l / WS, c = l - r * WS;
        int oh = wh7 + r + SHIFT; if (oh >= HG) oh -= HG;
        int ow = ww7 + c + SHIFT; if (ow >= HG) ow -= HG;
        out[(((size_t)nb * HG + oh) * HG + ow) * DIM + n] = acc[j] + bias;
      }
    }
  }
}

extern "C" void kernel_launch(void* const* d_in, const int* in_sizes, int n_in,
                              void* d_out, int out_size, void* d_ws, size_t ws_size,
                              hipStream_t stream) {
  const float* x      = (const float*)d_in[0];
  const float* qkv_w  = (const float*)d_in[1];
  const float* qkv_b  = (const float*)d_in[2];
  const float* proj_w = (const float*)d_in[3];
  const float* proj_b = (const float*)d_in[4];
  const float* table  = (const float*)d_in[5];
  float* out = (float*)d_out;

  // Workspace layout: qkv_w bf16 (221184 B) | proj_w bf16 (73728 B) | bias f32 (57624 B)
  unsigned short* wqkv  = (unsigned short*)d_ws;
  unsigned short* wproj = wqkv + NQKV * DIM;
  float* bias_g = (float*)((char*)d_ws + (size_t)(NQKV * DIM + DIM * DIM) * 2);

  prep_weights<<<(NQKV * DIM + 255) / 256, 256, 0, stream>>>(qkv_w, proj_w, wqkv, wproj);
  prep_bias<<<(NH * LTOK * LTOK + 255) / 256, 256, 0, stream>>>(table, bias_g);

  const int N = in_sizes[0] / (HG * HG * DIM);   // 32
  const int nBlocks = N * 64;                    // one window per workgroup
  const size_t smemBytes = 122880;               // 120 KB of the 320 KB WGP LDS
  swin_window_attn<<<nBlocks, 256, smemBytes, stream>>>(x, wqkv, qkv_b, wproj, proj_b, bias_g, out);
}
